// LSTMSeq2Seq_22892175688041
// MI455X (gfx1250) — matrix operand order repaired and checked by reading, not compiled, not run on hardware
//
#include <hip/hip_runtime.h>
#include <hip/hip_bf16.h>
#include <math.h>

// ---------------- problem constants ----------------
#define VSZ 32000
#define EMB 256
#define HID 512
#define BB  32
#define SS  64
#define TT  64
#define G4  (4*HID)   // 2048

// ---------------- WMMA types -----------------------
typedef __attribute__((ext_vector_type(16))) __bf16 v16bf;
typedef __attribute__((ext_vector_type(8)))  float  v8f;

union Frag { unsigned int u[8]; v16bf v; };

static __device__ __forceinline__ v8f wmma_bf16(const Frag& a, const Frag& b, v8f acc) {
    return __builtin_amdgcn_wmma_f32_16x16x32_bf16(false, a.v, false, b.v,
                                                   (short)0, acc, false, false);
}

static __device__ __forceinline__ unsigned short f32_to_bf16(float f) {
    unsigned int u = __float_as_uint(f);
    u += 0x7FFFu + ((u >> 16) & 1u);       // round-to-nearest-even
    return (unsigned short)(u >> 16);
}

static __device__ __forceinline__ float sigf(float x) { return 1.0f / (1.0f + expf(-x)); }

// A fragment: 16x32 bf16 tile, A row-major with leading dim lda (in elements),
// pointer already offset to (row0, k0).  ISA 7.12.2 16-bit A layout.
// Per lane this is two contiguous 16B chunks -> compiler emits 2x global_load_b128.
static __device__ __forceinline__ void load_a_frag(const unsigned short* A, int lda, Frag& fr) {
    int lane = threadIdx.x & 31;
    int m  = lane & 15;
    int hi = (lane >> 4) & 1;
    const unsigned short* p = A + (size_t)m * lda;
#pragma unroll
    for (int j = 0; j < 8; ++j) {
        int k = ((j >> 2) << 4) + hi * 8 + ((j & 3) << 1);   // K pairs per ISA layout
        fr.u[j] = *(const unsigned int*)(p + k);
    }
}

// B fragment: 32x16 bf16 (KxN), built from row-major W (N x K): B(k,n) = W[n][k].
// Pointer already offset to (n0, k0).  Lanes 0-15: K 0..7,16..23, lanes 16-31: K 8..15,24..31.
static __device__ __forceinline__ void load_b_frag(const unsigned short* W, int ldw, Frag& fr) {
    int lane = threadIdx.x & 31;
    int n  = lane & 15;
    int hi = (lane >> 4) & 1;
    const unsigned short* p = W + (size_t)n * ldw + hi * 8;
#pragma unroll
    for (int j = 0; j < 8; ++j) {
        fr.u[j] = *(const unsigned int*)(p + ((j >> 2) << 4) + ((j & 3) << 1));
    }
}

// ---------------- utility kernels ------------------
__global__ void k_conv_bf16(const float* __restrict__ src, unsigned short* __restrict__ dst, size_t n) {
    size_t i = (size_t)blockIdx.x * blockDim.x + threadIdx.x;
    size_t stride = (size_t)gridDim.x * blockDim.x;
    for (; i < n; i += stride) dst[i] = f32_to_bf16(src[i]);
}

__global__ void k_zero_f32(float* __restrict__ p, size_t n) {
    size_t i = (size_t)blockIdx.x * blockDim.x + threadIdx.x;
    size_t stride = (size_t)gridDim.x * blockDim.x;
    for (; i < n; i += stride) p[i] = 0.0f;
}

// zero d_out[:, 0, :]
__global__ void k_zero_first(float* __restrict__ out) {
    size_t i = (size_t)blockIdx.x * blockDim.x + threadIdx.x;
    if (i >= (size_t)BB * VSZ) return;
    size_t b = i / VSZ, n = i % VSZ;
    out[b * (size_t)TT * VSZ + n] = 0.0f;
}

// encoder embedding gather -> bf16 (B*S, EMB)
__global__ void k_gather_emb(const int* __restrict__ src, const float* __restrict__ emb,
                             unsigned short* __restrict__ out) {
    size_t i = (size_t)blockIdx.x * blockDim.x + threadIdx.x;
    if (i >= (size_t)BB * SS * EMB) return;
    size_t r = i >> 8;            // /EMB
    size_t e = i & (EMB - 1);
    out[i] = f32_to_bf16(emb[(size_t)src[r] * EMB + e]);
}

// dW_cat[n][0:768] = dWih[n], dW_cat[n][768:1280] = dWhh[n]  (bf16)
__global__ void k_build_dwcat(const float* __restrict__ dWih, const float* __restrict__ dWhh,
                              unsigned short* __restrict__ dst) {
    size_t i = (size_t)blockIdx.x * blockDim.x + threadIdx.x;
    size_t total = (size_t)G4 * (EMB + HID + HID);   // 2048*1280
    if (i >= total) return;
    size_t n = i / 1280, k = i % 1280;
    float v = (k < 768) ? dWih[n * 768 + k] : dWhh[n * 512 + (k - 768)];
    dst[i] = f32_to_bf16(v);
}

// enc_score[r] = dot(enc_out[r,:], attnW[0,512:1024])
__global__ void k_escore(const float* __restrict__ enc_out, const float* __restrict__ attnW,
                         float* __restrict__ enc_score) {
    int r = blockIdx.x * blockDim.x + threadIdx.x;
    if (r >= BB * SS) return;
    const float* p  = enc_out + (size_t)r * HID;
    const float* w2 = attnW + HID;
    float acc = 0.0f;
    for (int k = 0; k < HID; ++k) acc += p[k] * w2[k];
    enc_score[r] = acc;
}

// ---------------- generic WMMA GEMM (2x2 tiles / wave) ----------------
// C[M,N] = act( A[M,K](bf16) @ W[N,K](bf16)^T + bias[N] )
// Each wave owns a 32x32 macro-tile: 4 accumulators, a0/a1/b0/b1 fragments reused.
// Requires M % 32 == 0, N % 32 == 0, K % 32 == 0 (true for every call site).
__global__ void k_gemm_bf16(const unsigned short* __restrict__ A, int lda,
                            const unsigned short* __restrict__ W, int ldw,
                            const float* __restrict__ bias,
                            float* __restrict__ C, long long ldc,
                            int M, int N, int K, int act) {
    int wid = (int)((blockIdx.x * blockDim.x + threadIdx.x) >> 5);
    int pm = M >> 5, pn = N >> 5;
    if (wid >= pm * pn) return;
    int mp = wid % pm, np = wid / pm;
    int lane = threadIdx.x & 31;
    int hi = (lane >> 4) & 1;

    v8f acc[2][2];
#pragma unroll
    for (int i = 0; i < 2; ++i)
#pragma unroll
        for (int j = 0; j < 2; ++j)
            acc[i][j] = (v8f){0.f,0.f,0.f,0.f,0.f,0.f,0.f,0.f};

    const unsigned short* Ab0 = A + (size_t)mp * 32 * lda;
    const unsigned short* Ab1 = Ab0 + (size_t)16 * lda;
    const unsigned short* Wb0 = W + (size_t)np * 32 * ldw;
    const unsigned short* Wb1 = Wb0 + (size_t)16 * ldw;
    for (int k0 = 0; k0 < K; k0 += 32) {
        Frag a0, a1, b0, b1;
        load_a_frag(Ab0 + k0, lda, a0);
        load_a_frag(Ab1 + k0, lda, a1);
        load_b_frag(Wb0 + k0, ldw, b0);
        load_b_frag(Wb1 + k0, ldw, b1);
        acc[0][0] = wmma_bf16(a0, b0, acc[0][0]);
        acc[0][1] = wmma_bf16(a0, b1, acc[0][1]);
        acc[1][0] = wmma_bf16(a1, b0, acc[1][0]);
        acc[1][1] = wmma_bf16(a1, b1, acc[1][1]);
    }
#pragma unroll
    for (int j = 0; j < 2; ++j) {
        int n = np * 32 + j * 16 + (lane & 15);
        float bv = bias ? bias[n] : 0.0f;
#pragma unroll
        for (int i = 0; i < 2; ++i) {
#pragma unroll
            for (int r = 0; r < 8; ++r) {
                int m = mp * 32 + i * 16 + r + hi * 8;
                float v = acc[i][j][r] + bv;
                if (act == 1) v = tanhf(v);
                C[(size_t)m * (size_t)ldc + n] = v;
            }
        }
    }
}

// ---------------- encoder LSTM scan ----------------
// Single persistent workgroup, 32 waves. h double-buffered in LDS (bf16), c in regs.
// gates_pre (B*S, 2048) fp32 already includes x@Wih.T + b.
__global__ __launch_bounds__(1024)
void k_lstm_enc(const float* __restrict__ gates_pre,
                const unsigned short* __restrict__ Whh,      // (2048, 512) bf16
                unsigned short* __restrict__ enc_cat,        // (B*S, 1024) bf16
                unsigned short* __restrict__ hT_cat,         // (B, 1024) bf16
                int colofs, int dir) {
    __shared__ unsigned short h_lds[2][BB * HID];            // 2 * 32KB
    int tid = threadIdx.x;
    for (int i = tid; i < BB * HID; i += 1024) h_lds[0][i] = 0;
    __syncthreads();

    int wave = tid >> 5, lane = tid & 31;
    int hi = (lane >> 4) & 1;
    float c_reg[2][8];
#pragma unroll
    for (int g = 0; g < 2; ++g)
#pragma unroll
        for (int r = 0; r < 8; ++r) c_reg[g][r] = 0.0f;

    int cur = 0;
    for (int step = 0; step < SS; ++step) {
        int s = dir ? (SS - 1 - step) : step;
#pragma unroll 1
        for (int g = 0; g < 2; ++g) {
            int grp = wave * 2 + g;          // 64 groups: (m_tile, h_tile)
            int mt = grp & 1, ht = grp >> 1;
            v8f acc[4];
#pragma unroll
            for (int q = 0; q < 4; ++q) acc[q] = (v8f){0.f,0.f,0.f,0.f,0.f,0.f,0.f,0.f};

            const unsigned short* Ab = &h_lds[cur][mt * 16 * HID];
            for (int kk = 0; kk < HID; kk += 32) {
                Frag a;
                load_a_frag(Ab + kk, HID, a);
#pragma unroll
                for (int gate = 0; gate < 4; ++gate) {
                    Frag b;
                    load_b_frag(Whh + (size_t)(gate * HID + ht * 16) * HID + kk, HID, b);
                    acc[gate] = wmma_bf16(a, b, acc[gate]);
                }
            }
            int hidx = ht * 16 + (lane & 15);
#pragma unroll
            for (int r = 0; r < 8; ++r) {
                int brow = mt * 16 + r + hi * 8;                 // batch index
                size_t base = ((size_t)brow * SS + s) * (size_t)G4;
                float iv = acc[0][r] + gates_pre[base + hidx];
                float fv = acc[1][r] + gates_pre[base + HID + hidx];
                float gv = acc[2][r] + gates_pre[base + 2 * HID + hidx];
                float ov = acc[3][r] + gates_pre[base + 3 * HID + hidx];
                float c = sigf(fv) * c_reg[g][r] + sigf(iv) * tanhf(gv);
                c_reg[g][r] = c;
                float h = sigf(ov) * tanhf(c);
                unsigned short hb = f32_to_bf16(h);
                h_lds[cur ^ 1][brow * HID + hidx] = hb;
                enc_cat[((size_t)brow * SS + s) * 1024 + colofs + hidx] = hb;
                if (step == SS - 1) hT_cat[(size_t)brow * 1024 + colofs + hidx] = hb;
            }
        }
        __syncthreads();
        cur ^= 1;
    }
}

// ---------------- decoder attention ----------------
// one block per batch row; builds A_cat[b] = [ emb(256) | ctx(512) | h(512) ] in bf16
__global__ void k_attn(const float* __restrict__ h_buf, const float* __restrict__ enc_out,
                       const float* __restrict__ enc_score, const float* __restrict__ attnW,
                       const float* __restrict__ attnb, const float* __restrict__ dec_emb,
                       const int* __restrict__ tgt, int step,
                       unsigned short* __restrict__ A_cat) {
    __shared__ float red[256];
    __shared__ float wgt[SS];
    int b = blockIdx.x, tid = threadIdx.x;

    float s0 = 0.0f;
    for (int k = tid; k < HID; k += 256) s0 += h_buf[b * HID + k] * attnW[k];
    red[tid] = s0; __syncthreads();
    for (int off = 128; off > 0; off >>= 1) {
        if (tid < off) red[tid] += red[tid + off];
        __syncthreads();
    }
    if (tid < SS) wgt[tid] = red[0] + enc_score[b * SS + tid] + attnb[0];
    __syncthreads();
    if (tid == 0) {
        float mx = wgt[0];
        for (int i = 1; i < SS; ++i) mx = fmaxf(mx, wgt[i]);
        float sm = 0.0f;
        for (int i = 0; i < SS; ++i) { wgt[i] = expf(wgt[i] - mx); sm += wgt[i]; }
        float inv = 1.0f / sm;
        for (int i = 0; i < SS; ++i) wgt[i] *= inv;
    }
    __syncthreads();
    for (int e = tid; e < HID; e += 256) {
        float acc = 0.0f;
        for (int s = 0; s < SS; ++s) acc += wgt[s] * enc_out[((size_t)b * SS + s) * HID + e];
        A_cat[b * 1280 + EMB + e] = f32_to_bf16(acc);
    }
    int tok = tgt[b * TT + step];
    for (int e = tid; e < EMB; e += 256)
        A_cat[b * 1280 + e] = f32_to_bf16(dec_emb[(size_t)tok * EMB + e]);
    for (int e = tid; e < HID; e += 256)
        A_cat[b * 1280 + EMB + HID + e] = f32_to_bf16(h_buf[b * HID + e]);
}

// ---------------- decoder LSTM cell ----------------
// g = A_cat(32x1280) @ dW_cat(2048x1280)^T + db ; LSTM update.  64 waves total.
__global__ void k_dec_cell(const unsigned short* __restrict__ A_cat,
                           const unsigned short* __restrict__ dW_cat,
                           const float* __restrict__ db,
                           float* __restrict__ c_buf, float* __restrict__ h_buf,
                           unsigned short* __restrict__ h_bf) {
    int wid = (int)((blockIdx.x * blockDim.x + threadIdx.x) >> 5);
    if (wid >= 64) return;
    int mt = wid & 1, ht = wid >> 1;
    int lane = threadIdx.x & 31;
    int hi = (lane >> 4) & 1;

    v8f acc[4];
#pragma unroll
    for (int q = 0; q < 4; ++q) acc[q] = (v8f){0.f,0.f,0.f,0.f,0.f,0.f,0.f,0.f};

    const unsigned short* Ab = A_cat + (size_t)mt * 16 * 1280;
    for (int k0 = 0; k0 < 1280; k0 += 32) {
        Frag a;
        load_a_frag(Ab + k0, 1280, a);
#pragma unroll
        for (int gate = 0; gate < 4; ++gate) {
            Frag b;
            load_b_frag(dW_cat + (size_t)(gate * HID + ht * 16) * 1280 + k0, 1280, b);
            acc[gate] = wmma_bf16(a, b, acc[gate]);
        }
    }
    int hidx = ht * 16 + (lane & 15);
#pragma unroll
    for (int r = 0; r < 8; ++r) {
        int brow = mt * 16 + r + hi * 8;
        float iv = acc[0][r] + db[hidx];
        float fv = acc[1][r] + db[HID + hidx];
        float gv = acc[2][r] + db[2 * HID + hidx];
        float ov = acc[3][r] + db[3 * HID + hidx];
        size_t ci = (size_t)brow * HID + hidx;
        float c = sigf(fv) * c_buf[ci] + sigf(iv) * tanhf(gv);
        c_buf[ci] = c;
        float h = sigf(ov) * tanhf(c);
        h_buf[ci] = h;
        h_bf[ci] = f32_to_bf16(h);
    }
}

// ---------------- host launcher --------------------
extern "C" void kernel_launch(void* const* d_in, const int* in_sizes, int n_in,
                              void* d_out, int out_size, void* d_ws, size_t ws_size,
                              hipStream_t stream) {
    const int*   src     = (const int*)  d_in[0];
    const int*   tgt     = (const int*)  d_in[1];
    const float* enc_emb = (const float*)d_in[2];
    const float* dec_emb = (const float*)d_in[3];
    const float* Wih_f   = (const float*)d_in[4];
    const float* Whh_f   = (const float*)d_in[5];
    const float* b_f     = (const float*)d_in[6];
    const float* Wih_b   = (const float*)d_in[7];
    const float* Whh_b   = (const float*)d_in[8];
    const float* b_b     = (const float*)d_in[9];
    const float* encW    = (const float*)d_in[10];
    const float* encb    = (const float*)d_in[11];
    const float* dWih    = (const float*)d_in[12];
    const float* dWhh    = (const float*)d_in[13];
    const float* db      = (const float*)d_in[14];
    const float* attnW   = (const float*)d_in[15];
    const float* attnb   = (const float*)d_in[16];
    const float* projW   = (const float*)d_in[17];
    const float* projb   = (const float*)d_in[18];
    float* out = (float*)d_out;

    // workspace carve-up
    char* ws = (char*)d_ws;
    size_t off = 0;
    auto alloc = [&](size_t bytes) -> void* {
        void* p = ws + off;
        off = (off + bytes + 255) & ~(size_t)255;
        return p;
    };
    unsigned short* Wih_f_bf = (unsigned short*)alloc((size_t)G4 * EMB * 2);
    unsigned short* Whh_f_bf = (unsigned short*)alloc((size_t)G4 * HID * 2);
    unsigned short* Wih_b_bf = (unsigned short*)alloc((size_t)G4 * EMB * 2);
    unsigned short* Whh_b_bf = (unsigned short*)alloc((size_t)G4 * HID * 2);
    unsigned short* encW_bf  = (unsigned short*)alloc((size_t)HID * 1024 * 2);
    unsigned short* dW_cat   = (unsigned short*)alloc((size_t)G4 * 1280 * 2);
    unsigned short* projW_bf = (unsigned short*)alloc((size_t)VSZ * HID * 2);
    unsigned short* emb_bf   = (unsigned short*)alloc((size_t)BB * SS * EMB * 2);
    float*          gates_f  = (float*)alloc((size_t)BB * SS * G4 * 4);
    float*          gates_b  = (float*)alloc((size_t)BB * SS * G4 * 4);
    unsigned short* enc_cat  = (unsigned short*)alloc((size_t)BB * SS * 1024 * 2);
    unsigned short* hT_cat   = (unsigned short*)alloc((size_t)BB * 1024 * 2);
    float*          enc_out  = (float*)alloc((size_t)BB * SS * HID * 4);
    float*          enc_sc   = (float*)alloc((size_t)BB * SS * 4);
    float*          h_buf    = (float*)alloc((size_t)BB * HID * 4);
    float*          c_buf    = (float*)alloc((size_t)BB * HID * 4);
    unsigned short* h_bf     = (unsigned short*)alloc((size_t)BB * HID * 2);
    unsigned short* A_cat    = (unsigned short*)alloc((size_t)BB * 1280 * 2);
    (void)ws_size; (void)n_in; (void)in_sizes; (void)out_size;

    auto cvt = [&](const float* s, unsigned short* d, size_t n) {
        int blocks = (int)((n + 255) / 256);
        k_conv_bf16<<<blocks, 256, 0, stream>>>(s, d, n);
    };

    // --- one-time conversions ---
    cvt(Wih_f, Wih_f_bf, (size_t)G4 * EMB);
    cvt(Whh_f, Whh_f_bf, (size_t)G4 * HID);
    cvt(Wih_b, Wih_b_bf, (size_t)G4 * EMB);
    cvt(Whh_b, Whh_b_bf, (size_t)G4 * HID);
    cvt(encW,  encW_bf,  (size_t)HID * 1024);
    cvt(projW, projW_bf, (size_t)VSZ * HID);
    {
        size_t n = (size_t)G4 * 1280;
        k_build_dwcat<<<(int)((n + 255) / 256), 256, 0, stream>>>(dWih, dWhh, dW_cat);
    }
    {
        size_t n = (size_t)BB * SS * EMB;
        k_gather_emb<<<(int)((n + 255) / 256), 256, 0, stream>>>(src, enc_emb, emb_bf);
    }

    auto gemm = [&](const unsigned short* A, int lda, const unsigned short* W, int ldw,
                    const float* bias, float* C, long long ldc, int M, int N, int K, int act) {
        int waves = (M >> 5) * (N >> 5);      // one 32x32 macro-tile per wave
        int blocks = (waves * 32 + 255) / 256;
        k_gemm_bf16<<<blocks, 256, 0, stream>>>(A, lda, W, ldw, bias, C, ldc, M, N, K, act);
    };

    // --- encoder input projections (hoisted out of the scan) ---
    gemm(emb_bf, EMB, Wih_f_bf, EMB, b_f, gates_f, G4, BB * SS, G4, EMB, 0);
    gemm(emb_bf, EMB, Wih_b_bf, EMB, b_b, gates_b, G4, BB * SS, G4, EMB, 0);

    // --- bidirectional recurrence (persistent single-WG kernels) ---
    k_lstm_enc<<<1, 1024, 0, stream>>>(gates_f, Whh_f_bf, enc_cat, hT_cat, 0, 0);
    k_lstm_enc<<<1, 1024, 0, stream>>>(gates_b, Whh_b_bf, enc_cat, hT_cat, HID, 1);

    // --- encoder output projection + attention precompute ---
    gemm(enc_cat, 1024, encW_bf, 1024, encb, enc_out, HID, BB * SS, HID, 1024, 0);
    k_escore<<<(BB * SS + 255) / 256, 256, 0, stream>>>(enc_out, attnW, enc_sc);

    // --- decoder init: h0 = tanh([hT_f|hT_b] @ encW^T + encb), c0 = 0 ---
    gemm(hT_cat, 1024, encW_bf, 1024, encb, h_buf, HID, BB, HID, 1024, 1);
    k_zero_f32<<<((BB * HID) + 255) / 256, 256, 0, stream>>>(c_buf, (size_t)BB * HID);
    k_zero_first<<<((size_t)BB * VSZ + 255) / 256, 256, 0, stream>>>(out);

    // --- decoder scan ---
    for (int t = 0; t < TT - 1; ++t) {
        k_attn<<<BB, 256, 0, stream>>>(h_buf, enc_out, enc_sc, attnW, attnb,
                                       dec_emb, tgt, t, A_cat);
        k_dec_cell<<<16, 128, 0, stream>>>(A_cat, dW_cat, db, c_buf, h_buf, h_bf);
        // logits -> out[:, t+1, :]
        gemm(h_bf, HID, projW_bf, HID, projb,
             out + (size_t)(t + 1) * VSZ, (long long)TT * VSZ,
             BB, VSZ, HID, 0);
    }
}
